// DynamicHyperNetSIR_41566693490893
// MI455X (gfx1250) — compile-verified, matrix operand order checked by reading, not run on hardware
//
#include <hip/hip_runtime.h>
#include <hip/hip_bf16.h>
#include <stdint.h>

#define NN 10000   // nodes
#define EE 1000    // hyperedges
#define SS 30      // steps
#define KCPB 39    // full 32-wide K chunks per pass-1 block (39*32 = 1248)

typedef __attribute__((ext_vector_type(16))) __bf16 v16bf;
typedef __attribute__((ext_vector_type(8)))  __bf16 v8bf;
typedef __attribute__((ext_vector_type(8)))  float  v8f;

// Stateless uniform in [0,1): deterministic across graph replays.
__device__ __forceinline__ float u01(uint32_t a) {
  a ^= a >> 16; a *= 0x7feb352du;
  a ^= a >> 15; a *= 0x846ca68bu;
  a ^= a >> 16;
  return (float)(a >> 8) * (1.0f / 16777216.0f);
}

__device__ __forceinline__ float clamp01(float x) {
  return fminf(1.0f, fmaxf(0.0f, x));
}

// Build bf16 A fragment from two contiguous 8-float runs (4x float4).
__device__ __forceinline__ v16bf pack16(const float4& q0, const float4& q1,
                                        const float4& q2, const float4& q3) {
  v16bf a;
  a[0]  = (__bf16)q0.x; a[1]  = (__bf16)q0.y; a[2]  = (__bf16)q0.z; a[3]  = (__bf16)q0.w;
  a[4]  = (__bf16)q1.x; a[5]  = (__bf16)q1.y; a[6]  = (__bf16)q1.z; a[7]  = (__bf16)q1.w;
  a[8]  = (__bf16)q2.x; a[9]  = (__bf16)q2.y; a[10] = (__bf16)q2.z; a[11] = (__bf16)q2.w;
  a[12] = (__bf16)q3.x; a[13] = (__bf16)q3.y; a[14] = (__bf16)q3.z; a[15] = (__bf16)q3.w;
  return a;
}

__device__ __forceinline__ v16bf join8(const v8bf& lo, const v8bf& hi) {
  v16bf r;
#pragma unroll
  for (int i = 0; i < 8; ++i) { r[i] = lo[i]; r[8 + i] = hi[i]; }
  return r;
}

// pathogen[0] = x ; state[0] = x
__global__ void __launch_bounds__(256) k_init(const float* __restrict__ x,
                                              float* __restrict__ pathogen,
                                              float* __restrict__ state) {
  int i = blockIdx.x * 256 + threadIdx.x;
  if (i < NN * 3) {
    float v = x[i];
    pathogen[i] = v;
    state[i] = v;
  }
}

// v[i] = state_t[i][1]; also zero the y accumulator for pass 1's atomics.
__global__ void __launch_bounds__(256) k_extract(const float* __restrict__ s_t,
                                                 float* __restrict__ v,
                                                 float* __restrict__ y) {
  int i = blockIdx.x * 256 + threadIdx.x;
  if (i < NN) v[i] = s_t[i * 3 + 1];
  if (i < EE) y[i] = 0.0f;
}

// Pass 1: y[e] += sum_{k in slice} H_t[e,k] * v[k] via v_wmma_f32_16x16x32_bf16.
// Grid (63 edge-tiles, 8 K-partitions) = 504 blocks / 4032 waves to cover HBM
// latency at 23.3 TB/s. Branch-free hot loop; exact-integer f32 atomics keep
// the result bitwise deterministic regardless of accumulation order.
__global__ void __launch_bounds__(256) k_matvec1(const float* __restrict__ Ht,
                                                 const float* __restrict__ v,
                                                 float* __restrict__ y) {
  // Block's 1248-wide K slice of v in bf16, + tail + zero page [1280..1311].
  __shared__ __attribute__((aligned(16))) __bf16 svb[1312];
  __shared__ float red[8][16];
  const int tid = threadIdx.x;
  const int kb0 = blockIdx.y * (KCPB * 32);       // slice base in K
  for (int i = tid; i < 1312; i += 256) {
    int gk = kb0 + i;
    svb[i] = ((i < 1264) && (gk < NN)) ? (__bf16)v[gk] : (__bf16)0.0f;
  }
  __syncthreads();

  const int wave = tid >> 5;
  const int lane = tid & 31;
  const int m0 = blockIdx.x * 16;                 // edge-row tile base
  const int M = lane & 15;                        // A row within tile
  const int h = lane >> 4;                        // lane half -> K sub-range
  const int row = min(m0 + M, EE - 1);            // clamp: dup rows discarded at store
  const float* rowp = Ht + (size_t)row * NN;
  const bool bl = (M == 0);                       // lanes 0/16 carry B column N=0

  v8f c = {};
  // ~5 chunks per wave; all loads unconditional.
  for (int ci = wave; ci < KCPB; ci += 8) {
    const int rel = ci * 32 + 8 * h;              // offset within slice
    const float* p = rowp + kb0 + rel;
    __builtin_prefetch(p + 512);                  // global_prefetch_b8, 2 iters ahead
    float4 q0 = *(const float4*)(p + 0);
    float4 q1 = *(const float4*)(p + 4);
    float4 q2 = *(const float4*)(p + 16);
    float4 q3 = *(const float4*)(p + 20);
    uint32_t off = bl ? (uint32_t)rel : 1280u;    // zero page for non-carriers
    v8bf blo = *(const v8bf*)(svb + off);
    v8bf bhi = *(const v8bf*)(svb + off + 16);
    v16bf a = pack16(q0, q1, q2, q3);
    v16bf b = join8(blo, bhi);
    c = __builtin_amdgcn_wmma_f32_16x16x32_bf16(false, a, false, b,
                                                (short)0, c, false, false);
  }
  // Global tail chunk #312 (K = 9984..9999): partition 7, wave 7 only.
  if (blockIdx.y == 7 && wave == 7) {
    const int rel = KCPB * 32 + 8 * h;            // 1248+8h -> K 9984..9999
    const float* p = rowp + kb0 + rel;            // in-row, no OOB
    float4 q0 = *(const float4*)(p + 0);
    float4 q1 = *(const float4*)(p + 4);
    float4 z4 = make_float4(0.f, 0.f, 0.f, 0.f);
    uint32_t off = bl ? (uint32_t)rel : 1280u;
    v8bf blo = *(const v8bf*)(svb + off);         // svb[1264..] are zero
    v8bf bhi = *(const v8bf*)(svb + off + 16);
    v16bf a = pack16(q0, q1, z4, z4);             // upper K half structurally zero
    v16bf b = join8(blo, bhi);
    c = __builtin_amdgcn_wmma_f32_16x16x32_bf16(false, a, false, b,
                                                (short)0, c, false, false);
  }
  // Column N=0 of C: lane 0 holds rows 0..7 (VGPR r), lane 16 holds rows 8..15.
  if (lane == 0) {
#pragma unroll
    for (int r = 0; r < 8; ++r) red[wave][r] = c[r];
  } else if (lane == 16) {
#pragma unroll
    for (int r = 0; r < 8; ++r) red[wave][8 + r] = c[r];
  }
  __syncthreads();
  if (tid < 16 && (m0 + tid) < EE) {
    float s = 0.f;
#pragma unroll
    for (int w = 0; w < 8; ++w) s += red[w][tid];
    // Exact small-integer f32 adds: order-independent, deterministic.
    __hip_atomic_fetch_add(&y[m0 + tid], s, __ATOMIC_RELAXED,
                           __HIP_MEMORY_SCOPE_AGENT);
  }
}

// Pass 2: z[n] = sum_e H_t[e,n] * y[e] via WMMA (A row 0 = y chunk from bf16 LDS,
// B = 32x16 H tile, rows clamped; zero-padded A kills out-of-range E columns),
// fused with the per-node SIR probability/state update. 625 blocks, L2-resident H.
__global__ void __launch_bounds__(256) k_matvec2_update(
    const float* __restrict__ Ht, const float* __restrict__ y,
    const float* __restrict__ p_prev, const float* __restrict__ s_prev,
    const float* __restrict__ beta, const float* __restrict__ gamma,
    float* __restrict__ p_out, float* __restrict__ s_out, int t) {
  __shared__ __attribute__((aligned(16))) __bf16 syb[1056];  // [1000..1055] = 0
  __shared__ float red[8][16];
  const int tid = threadIdx.x;
  for (int i = tid; i < 1056; i += 256)
    syb[i] = (i < EE) ? (__bf16)y[i] : (__bf16)0.0f;
  __syncthreads();

  const int wave = tid >> 5;
  const int lane = tid & 31;
  const int n0 = blockIdx.x * 16;                 // node tile base (10000/16 exact)
  const int nl = lane & 15;                       // B column (node) within tile
  const int h = lane >> 4;
  const bool al = (nl == 0);                      // lanes 0/16 carry A row M=0 (= y)
  const float* colp = Ht + n0 + nl;

  v8f c = {};
  // 32 chunks of 32 edges (e0 = 0..992), exactly 4 per wave; rows clamped to 999,
  // out-of-range K contributions are zero because syb[k>=1000] == 0.
  for (int e0 = wave * 32; e0 < EE; e0 += 256) {
    uint32_t aoff = al ? (uint32_t)(e0 + 8 * h) : 1024u;  // zero page for non-carriers
    v8bf alo = *(const v8bf*)(syb + aoff);
    v8bf ahi = *(const v8bf*)(syb + aoff + 16);
    v16bf a = join8(alo, ahi);
    v16bf b;
#pragma unroll
    for (int j = 0; j < 8; ++j) {
      int kbase = ((j < 4) ? (2 * j) : (16 + 2 * (j - 4))) + 8 * h;
      uint32_t e0c = (uint32_t)min(e0 + kbase, EE - 1);
      uint32_t e1c = (uint32_t)min(e0 + kbase + 1, EE - 1);
      float b0 = colp[e0c * (uint32_t)NN];
      float b1 = colp[e1c * (uint32_t)NN];
      b[2 * j]     = (__bf16)b0;
      b[2 * j + 1] = (__bf16)b1;
    }
    c = __builtin_amdgcn_wmma_f32_16x16x32_bf16(false, a, false, b,
                                                (short)0, c, false, false);
  }
  // Row M=0 of C: VGPR 0, lanes 0..15 hold N=0..15.
  if (lane < 16) red[wave][lane] = c[0];
  __syncthreads();

  if (tid < 16) {
    const int node = n0 + tid;
    float z = 0.f;
#pragma unroll
    for (int w = 0; w < 8; ++w) z += red[w][tid];

    const float inf = s_prev[node * 3 + 1];
    const float new_cases = beta[node] * z;
    const float new_rec = gamma[node] * inf;
    const float p0 = clamp01(p_prev[node * 3 + 0] - new_cases);
    const float p1 = clamp01(p_prev[node * 3 + 1] + new_cases - new_rec);
    const float p2 = clamp01(p_prev[node * 3 + 2] + new_rec);

    const uint32_t base = (uint32_t)node * 2654435761u ^ ((uint32_t)t * 0x9E3779B9u);
    const float u1 = u01(base ^ 0x1234567u);
    const float u2 = u01(base ^ 0x89ABCDEu);

    const bool wasS = (s_prev[node * 3 + 0] == 1.0f);
    const bool wasI = (inf == 1.0f);
    const bool s_to_I = wasS && (u1 < p1);
    const bool i_event = wasI && (u1 < p2);
    const bool i_to_R = i_event && (u2 < 0.5f);
    const bool i_to_S = i_event && !(u2 < 0.5f);
    const bool nS = (wasS && !s_to_I) || i_to_S;
    const bool nI = s_to_I || (wasI && !i_event);
    const bool nR = (!wasS && !wasI) || i_to_R;

    p_out[node * 3 + 0] = p0;
    p_out[node * 3 + 1] = p1;
    p_out[node * 3 + 2] = p2;
    s_out[node * 3 + 0] = nS ? 1.0f : 0.0f;
    s_out[node * 3 + 1] = nI ? 1.0f : 0.0f;
    s_out[node * 3 + 2] = nR ? 1.0f : 0.0f;
  }
}

extern "C" void kernel_launch(void* const* d_in, const int* in_sizes, int n_in,
                              void* d_out, int out_size, void* d_ws, size_t ws_size,
                              hipStream_t stream) {
  const float* x     = (const float*)d_in[0];  // (N,3)
  const float* H     = (const float*)d_in[1];  // (S-1, E, N)
  const float* beta  = (const float*)d_in[2];  // (N,)
  const float* gamma = (const float*)d_in[3];  // (N,)
  (void)in_sizes; (void)n_in; (void)out_size; (void)ws_size;

  float* pathogen = (float*)d_out;                       // (S, N, 3)
  float* state    = pathogen + (size_t)SS * NN * 3;      // (S, N, 3)
  float* ws_v = (float*)d_ws;                            // N floats
  float* ws_y = ws_v + NN;                               // E floats

  k_init<<<(NN * 3 + 255) / 256, 256, 0, stream>>>(x, pathogen, state);

  for (int t = 0; t < SS - 1; ++t) {
    const float* Ht = H + (size_t)t * EE * NN;
    const float* p_prev = pathogen + (size_t)t * NN * 3;
    const float* s_prev = state + (size_t)t * NN * 3;
    float* p_next = pathogen + (size_t)(t + 1) * NN * 3;
    float* s_next = state + (size_t)(t + 1) * NN * 3;

    k_extract<<<(NN + 255) / 256, 256, 0, stream>>>(s_prev, ws_v, ws_y);
    k_matvec1<<<dim3((EE + 15) / 16, 8), 256, 0, stream>>>(Ht, ws_v, ws_y);
    k_matvec2_update<<<NN / 16, 256, 0, stream>>>(Ht, ws_y, p_prev, s_prev,
                                                  beta, gamma, p_next, s_next, t);
  }
}
